// MultiQueryAttention_19310172962885
// MI455X (gfx1250) — matrix operand order repaired and checked by reading, not compiled, not run on hardware
//
#include <hip/hip_runtime.h>
#include <hip/hip_bf16.h>

// ---------------------------------------------------------------------------
// MQA forward returning attn_weights only (V path is dead code in reference).
// B=2, H=16, S=2048, D_MODEL=1024, D_HEAD=64.
// Output-bandwidth bound (512MB fp32 attn_weights @ 23.3 TB/s ~= 23us floor),
// so: f16 WMMA (f32 accum) for all GEMMs, scores computed ONCE into 128KB LDS,
// softmax on-chip, single coalesced NON-TEMPORAL b128 stream of the output.
// ---------------------------------------------------------------------------

typedef __attribute__((ext_vector_type(16))) _Float16 v16h;
typedef __attribute__((ext_vector_type(8)))  float    v8f;
typedef __attribute__((ext_vector_type(4)))  float    f32x4;

#define S_LEN   2048
#define DHEAD   64
#define NHEADS  16
#define LSTR    2052   // 2048 + 4 pad: 8-row delta shifts banks by 32 -> no
                       // conflict between the two half-wave store groups;
                       // row stride = 513*16B keeps float4 alignment.

// ---------------------------------------------------------------------------
// Projection: out = f16( X[4096x1024] @ W[Nx1024]^T + bias ), stored as
// out[((b*H + n>>6)*2048 + s)*64 + (n&63)]  (head-major so attention reads
// contiguous 64-half rows).  One wave computes one 16x16 tile; K-loop by 32.
// ---------------------------------------------------------------------------
__global__ __launch_bounds__(256)
void proj_f16_kernel(const float* __restrict__ X, const float* __restrict__ W,
                     const float* __restrict__ bias, _Float16* __restrict__ out,
                     int N, int H, int wavesPerBlock)
{
    const int lane = threadIdx.x & 31;
    const int wave = threadIdx.x >> 5;
    const int m0 = blockIdx.x * 16;
    const int n0 = (blockIdx.y * wavesPerBlock + wave) * 16;
    if (n0 >= N) return;                      // wave-uniform: EXEC stays all-1s

    // A-fragment layout (f16 16x32): lane holds row = lane&15;
    // lanes<16 hold K {0..7,16..23}, lanes>=16 hold K {8..15,24..31}.
    const int  ksel = (lane & 16) ? 8 : 0;
    const int  row  = m0 + (lane & 15);
    const float* xrow = X + (size_t)row * 1024;

    // B-fragment layout (f16 32x16): lane holds col n = n0+(lane&15);
    // lanes<16 hold K {0..7,16..23}, lanes>=16 hold K {8..15,24..31}.
    const int  ncol = n0 + (lane & 15);
    const float* wrow = W + (size_t)ncol * 1024 + ((lane & 16) ? 8 : 0);

    v8f c = {};
    for (int k = 0; k < 1024; k += 32) {
        v16h a, bf;
#pragma unroll
        for (int e = 0; e < 8; ++e) {
            a[e]     = (_Float16)xrow[k + ksel + e];
            a[8 + e] = (_Float16)xrow[k + 16 + ksel + e];
        }
#pragma unroll
        for (int e = 0; e < 8; ++e) { bf[e] = (_Float16)wrow[k + e]; bf[8 + e] = (_Float16)wrow[k + 16 + e]; }
        c = __builtin_amdgcn_wmma_f32_16x16x32_f16(false, a, false, bf,
                                                   (short)0, c, false, false);
    }

    const float bv = bias[ncol];
    const int hh = ncol >> 6;       // head index (H=16) or 0 (H=1)
    const int dd = ncol & 63;
    const int rbase = (lane & 16) ? 8 : 0;   // D layout: reg r -> M = r(+8)
#pragma unroll
    for (int r = 0; r < 8; ++r) {
        const int m  = m0 + r + rbase;
        const int bb = m >> 11;              // batch (S=2048)
        const int ss = m & 2047;
        out[(((size_t)bb * H + hh) * S_LEN + ss) * DHEAD + dd] =
            (_Float16)(c[r] + bv);
    }
}

// ---------------------------------------------------------------------------
// Scores + softmax.  One block = one (b, h, 16-query-row block).
// 8 waves: each wave owns 256 key columns (16 tiles of 16x16).
// Scores (f32, pre-scaled by 1/8) land in LDS [16][LSTR]; softmax reduces
// per row with wave shuffles; exp written back in place; final pass streams
// out[row][col] = exp / sum as non-temporal b128 stores (no L2 pollution:
// the 512MB output is write-once, 2.7x the 192MB L2).
// ---------------------------------------------------------------------------
__global__ __launch_bounds__(256)
void mqa_scores_softmax(const _Float16* __restrict__ Qh,
                        const _Float16* __restrict__ Kh,
                        float* __restrict__ out)
{
    extern __shared__ float smem[];          // 16*LSTR scores + 16 rowInv
    float* rowInv = smem + 16 * LSTR;

    const int tid  = threadIdx.x;
    const int lane = tid & 31;
    const int wave = tid >> 5;

    int gid = blockIdx.x;                    // [0, 4096)
    const int qblk = gid & 127;  gid >>= 7;
    const int h    = gid & 15;   gid >>= 4;
    const int b    = gid;                    // 0..1
    const int q0   = qblk * 16;

    // --- load Q A-fragments once (d-split: a0 = d 0..31, a1 = d 32..63) ---
    const int ksel = (lane & 16) ? 8 : 0;
    const _Float16* qrow =
        Qh + (((size_t)(b * NHEADS + h)) * S_LEN + q0 + (lane & 15)) * DHEAD;
    v16h a0, a1;
#pragma unroll
    for (int e = 0; e < 8; ++e) {
        a0[e]     = qrow[ksel + e];
        a0[8 + e] = qrow[16 + ksel + e];
        a1[e]     = qrow[32 + ksel + e];
        a1[8 + e] = qrow[48 + ksel + e];
    }

    // --- score tiles: this wave covers key cols [wave*256, wave*256+256) ---
    const int bksel = (lane & 16) ? 8 : 0;   // B frag: lanes>=16 hold K {8..15,24..31}
    const int rbase = (lane & 16) ? 8 : 0;
    for (int t = 0; t < 16; ++t) {
        const int k0 = (wave * 16 + t) * 16;
        const _Float16* krow =
            Kh + ((size_t)b * S_LEN + (k0 + (lane & 15))) * DHEAD + bksel;
        v16h b0, b1;
#pragma unroll
        for (int e = 0; e < 8; ++e) { b0[e] = krow[e]; b0[8 + e] = krow[16 + e]; b1[e] = krow[32 + e]; b1[8 + e] = krow[48 + e]; }
        v8f c = {};
        c = __builtin_amdgcn_wmma_f32_16x16x32_f16(false, a0, false, b0,
                                                   (short)0, c, false, false);
        c = __builtin_amdgcn_wmma_f32_16x16x32_f16(false, a1, false, b1,
                                                   (short)0, c, false, false);
        const int col = k0 + (lane & 15);
#pragma unroll
        for (int r = 0; r < 8; ++r)
            smem[(rbase + r) * LSTR + col] = c[r] * 0.125f;   // 1/sqrt(64)
    }
    __syncthreads();

    // --- per-row softmax stats: wave w handles rows w and w+8 ---
#pragma unroll
    for (int rr = 0; rr < 2; ++rr) {
        const int r = wave + 8 * rr;
        float* rowp = smem + r * LSTR;
        float m = -3.402823466e38f;
#pragma unroll 4
        for (int i = 0; i < S_LEN / 32; ++i)
            m = fmaxf(m, rowp[lane + 32 * i]);
#pragma unroll
        for (int off = 16; off > 0; off >>= 1)
            m = fmaxf(m, __shfl_xor(m, off, 32));
        float s = 0.f;
#pragma unroll 4
        for (int i = 0; i < S_LEN / 32; ++i) {
            const float ev =
                exp2f((rowp[lane + 32 * i] - m) * 1.4426950408889634f);
            rowp[lane + 32 * i] = ev;
            s += ev;
        }
#pragma unroll
        for (int off = 16; off > 0; off >>= 1)
            s += __shfl_xor(s, off, 32);
        if (lane == 0) rowInv[r] = 1.0f / s;
    }
    __syncthreads();

    // --- streaming output: 32 iterations of b128 non-temporal stores ---
    // float4 units: 16*2048/4 = 8192 elems; per iteration the 256 threads
    // cover one aligned half-row (r is uniform within an iteration).
    f32x4* out4 = reinterpret_cast<f32x4*>(
        out + (((size_t)(b * NHEADS + h)) * S_LEN + q0) * (size_t)S_LEN);
    for (int e = tid; e < 16 * (S_LEN / 4); e += 256) {
        const int r  = e >> 9;               // 512 float4 per row
        const int c4 = e & 511;
        const f32x4 v =
            *reinterpret_cast<const f32x4*>(smem + r * LSTR + 4 * c4);
        const float inv = rowInv[r];
        __builtin_nontemporal_store(v * inv,
                                    out4 + (size_t)r * (S_LEN / 4) + c4);
    }
}

// ---------------------------------------------------------------------------
extern "C" void kernel_launch(void* const* d_in, const int* in_sizes, int n_in,
                              void* d_out, int out_size, void* d_ws, size_t ws_size,
                              hipStream_t stream)
{
    (void)in_sizes; (void)n_in; (void)out_size; (void)ws_size;
    const float* query = (const float*)d_in[0];
    const float* key   = (const float*)d_in[1];
    // d_in[2] (value), d_in[7], d_in[8] (Wv) are dead in the reference output.
    const float* Wq_w  = (const float*)d_in[3];
    const float* Wq_b  = (const float*)d_in[4];
    const float* Wk_w  = (const float*)d_in[5];
    const float* Wk_b  = (const float*)d_in[6];
    float* out = (float*)d_out;

    _Float16* Qh = (_Float16*)d_ws;                         // 2*16*2048*64 = 8MB
    _Float16* Kh = Qh + (size_t)2 * NHEADS * S_LEN * DHEAD; // +0.5MB

    // Q projection: M=4096, N=1024; 8 waves/block -> grid (256, 8)
    dim3 gq(4096 / 16, 1024 / (16 * 8));
    proj_f16_kernel<<<gq, 256, 0, stream>>>(query, Wq_w, Wq_b, Qh,
                                            1024, NHEADS, 8);
    // K projection: M=4096, N=64; 4 waves/block (128 threads) -> grid (256, 1)
    dim3 gk(4096 / 16, 1);
    proj_f16_kernel<<<gk, 128, 0, stream>>>(key, Wk_w, Wk_b, Kh,
                                            64, 1, 4);

    // Attention: 4096 blocks (b,h,qblk), 256 threads, ~128KB LDS
    const size_t shmem = (size_t)(16 * LSTR + 16) * sizeof(float);
    mqa_scores_softmax<<<4096, 256, shmem, stream>>>(Qh, Kh, out);
}
